// MoELayer_24945170055507
// MI455X (gfx1250) — matrix operand order repaired and checked by reading, not compiled, not run on hardware
//
#include <hip/hip_runtime.h>

// ---------------------------------------------------------------------------
// MoE (top-2 of 8 experts + shared expert), fused SwiGLU, bf16 WMMA GEMMs.
// T=4096 tokens, H=1024, E=8, top-2, I=1408, IS=2816.
//
// MI455X: ~355 GF of GEMM vs ~200MB operand traffic (~9us @ 23.3 TB/s)
// => compute bound => v_wmma_f32_16x16x32_bf16 with f32 accumulation.
// Per-expert fp32 weights (<=11.5MB per kernel) stay L2-resident (192MB L2);
// fp32->bf16 conversion happens during global->LDS staging.
//
// Register budget drives the pipelining choice per kernel:
//  - gateup: 128 acc VGPRs + 64 frag VGPRs -> single-buffered LDS (no spills;
//    latency hidden by 16 waves/WGP occupancy + compiler dscnt overlap).
//  - down:   64 acc VGPRs -> double-buffered LDS + register prefetch.
// ---------------------------------------------------------------------------

#define T_TOK 4096
#define HDIM  1024
#define NEXP  8
#define IDIM  1408
#define ISDIM 2816

typedef __attribute__((ext_vector_type(16))) __bf16 v16bf;
typedef __attribute__((ext_vector_type(8)))  float  v8f;

union Frag {
    v16bf v;
    uint4 q[2];
};

__device__ __forceinline__ unsigned int f2bf_bits(float f) {
    union { float f; unsigned u; } c; c.f = f;
    return (c.u + 0x7FFFu + ((c.u >> 16) & 1u)) >> 16;   // round-to-nearest-even
}
__device__ __forceinline__ unsigned short f2bf(float f) {
    return (unsigned short)f2bf_bits(f);
}
// pack bf16(lo), bf16(hi) into one dword (lo = smaller k, hi = k+1)
__device__ __forceinline__ unsigned int f2bf_pk(float lo, float hi) {
    return f2bf_bits(lo) | (f2bf_bits(hi) << 16);
}

__device__ __forceinline__ float fast_sigmoid(float x) {
    return __builtin_amdgcn_rcpf(1.0f + __expf(-x));     // v_rcp_f32, no div ladder
}

#define LDP 40   // padded LDS row stride in ushorts (80B rows, 16B aligned)

// ---------------------------------------------------------------------------
// Fused gate/up GEMM + SwiGLU:  Hout = silu(A@Bg) * (A@Bu)   (bf16 out)
// A: [T][K] bf16, Bg/Bu: [K][N] fp32, Hout: [T][N] bf16
// Grid: (T/128, N/128), 256 threads (8 wave32s, 2x4), 4x2 WMMA tiles per wave.
// Single-buffered LDS: 128 accumulator VGPRs leave no room for prefetch regs.
// ---------------------------------------------------------------------------
__global__ __launch_bounds__(256)
void moe_gateup_kernel(const unsigned short* __restrict__ A,
                       const float* __restrict__ Bg,
                       const float* __restrict__ Bu,
                       unsigned short* __restrict__ Hout,
                       int K, int N)
{
    __shared__ unsigned short lA [128 * LDP];
    __shared__ unsigned short lBg[128 * LDP];
    __shared__ unsigned short lBu[128 * LDP];

    const int tid   = threadIdx.x;
    const int lane  = tid & 31;
    const int wid   = tid >> 5;
    const int waveM = wid >> 2;      // 0..1  (64 rows each)
    const int waveN = wid & 3;       // 0..3  (32 cols each)
    const int half  = lane >> 4;
    const int l16   = lane & 15;

    const int row0 = blockIdx.x * 128;
    const int col0 = blockIdx.y * 128;

    v8f zero = {};
    v8f accG[4][2], accU[4][2];
#pragma unroll
    for (int i = 0; i < 4; ++i)
#pragma unroll
        for (int j = 0; j < 2; ++j) { accG[i][j] = zero; accU[i][j] = zero; }

    const int ksteps = K >> 5;
    for (int kt = 0; kt < ksteps; ++kt) {
        const int k0 = kt << 5;
        __syncthreads();
        // --- stage A tile: 128x32 bf16, b128 chunks ---
#pragma unroll
        for (int it = 0; it < 2; ++it) {
            int c  = tid + it * 256;
            int r  = c >> 2;
            int cc = (c & 3) << 3;
            uint4 v = *(const uint4*)(A + (size_t)(row0 + r) * K + k0 + cc);
            *(uint4*)&lA[r * LDP + cc] = v;
        }
        // --- stage B tiles: k-row pairs x 4 cols, packed b32 stores to [N][K] ---
#pragma unroll
        for (int it = 0; it < 2; ++it) {
            int c  = tid + it * 256;          // 512 chunks: 16 k-pairs x 32 col4
            int kk = (c >> 5) << 1;           // k = 0,2,..,30
            int cc = (c & 31) << 2;           // n col, x4
            const float4 g0 = *(const float4*)(Bg + (size_t)(k0 + kk) * N + col0 + cc);
            const float4 g1 = *(const float4*)(Bg + (size_t)(k0 + kk + 1) * N + col0 + cc);
            const float4 u0 = *(const float4*)(Bu + (size_t)(k0 + kk) * N + col0 + cc);
            const float4 u1 = *(const float4*)(Bu + (size_t)(k0 + kk + 1) * N + col0 + cc);
            *(unsigned int*)&lBg[(cc + 0) * LDP + kk] = f2bf_pk(g0.x, g1.x);
            *(unsigned int*)&lBg[(cc + 1) * LDP + kk] = f2bf_pk(g0.y, g1.y);
            *(unsigned int*)&lBg[(cc + 2) * LDP + kk] = f2bf_pk(g0.z, g1.z);
            *(unsigned int*)&lBg[(cc + 3) * LDP + kk] = f2bf_pk(g0.w, g1.w);
            *(unsigned int*)&lBu[(cc + 0) * LDP + kk] = f2bf_pk(u0.x, u1.x);
            *(unsigned int*)&lBu[(cc + 1) * LDP + kk] = f2bf_pk(u0.y, u1.y);
            *(unsigned int*)&lBu[(cc + 2) * LDP + kk] = f2bf_pk(u0.z, u1.z);
            *(unsigned int*)&lBu[(cc + 3) * LDP + kk] = f2bf_pk(u0.w, u1.w);
        }
        __syncthreads();

        // --- fragments (contiguous k = 16*half + s, same order for A and B) ---
        Frag fa[4];
#pragma unroll
        for (int mi = 0; mi < 4; ++mi) {
            int m = waveM * 64 + mi * 16 + l16;
            fa[mi].q[0] = *(const uint4*)&lA[m * LDP + 16 * half];
            fa[mi].q[1] = *(const uint4*)&lA[m * LDP + 16 * half + 8];
        }
        Frag fg[2], fu[2];
#pragma unroll
        for (int ni = 0; ni < 2; ++ni) {
            int n = waveN * 32 + ni * 16 + l16;
            fg[ni].q[0] = *(const uint4*)&lBg[n * LDP + 16 * half];
            fg[ni].q[1] = *(const uint4*)&lBg[n * LDP + 16 * half + 8];
            fu[ni].q[0] = *(const uint4*)&lBu[n * LDP + 16 * half];
            fu[ni].q[1] = *(const uint4*)&lBu[n * LDP + 16 * half + 8];
        }
#pragma unroll
        for (int mi = 0; mi < 4; ++mi)
#pragma unroll
            for (int ni = 0; ni < 2; ++ni) {
                accG[mi][ni] = __builtin_amdgcn_wmma_f32_16x16x32_bf16(
                    false, fa[mi].v, false, fg[ni].v, (short)0, accG[mi][ni], false, false);
                accU[mi][ni] = __builtin_amdgcn_wmma_f32_16x16x32_bf16(
                    false, fa[mi].v, false, fu[ni].v, (short)0, accU[mi][ni], false, false);
            }
    }

    // --- epilogue: silu(g) * u -> bf16 ---
#pragma unroll
    for (int mi = 0; mi < 4; ++mi)
#pragma unroll
        for (int ni = 0; ni < 2; ++ni)
#pragma unroll
            for (int r = 0; r < 8; ++r) {
                int m = row0 + waveM * 64 + mi * 16 + r + 8 * half;
                int n = col0 + waveN * 32 + ni * 16 + l16;
                float g = accG[mi][ni][r];
                float u = accU[mi][ni][r];
                Hout[(size_t)m * N + n] = f2bf(g * fast_sigmoid(g) * u);
            }
}

// ---------------------------------------------------------------------------
// Down GEMM with per-token scale, accumulated into fp32 out:
//   Out[t][n] (+)= scale[t] * (A @ B)[t][n]
// Double-buffered LDS + register prefetch (64 acc VGPRs -> headroom).
// ---------------------------------------------------------------------------
__global__ __launch_bounds__(256)
void moe_down_kernel(const unsigned short* __restrict__ A,
                     const float* __restrict__ B,
                     const float* __restrict__ scale,
                     float* __restrict__ Out,
                     int K, int N, int accumulate)
{
    __shared__ unsigned short lA[2][128 * LDP];
    __shared__ unsigned short lB[2][128 * LDP];

    const int tid   = threadIdx.x;
    const int lane  = tid & 31;
    const int wid   = tid >> 5;
    const int waveM = wid >> 2;
    const int waveN = wid & 3;
    const int half  = lane >> 4;
    const int l16   = lane & 15;

    const int row0 = blockIdx.x * 128;
    const int col0 = blockIdx.y * 128;

    v8f zero = {};
    v8f acc[4][2];
#pragma unroll
    for (int i = 0; i < 4; ++i)
#pragma unroll
        for (int j = 0; j < 2; ++j) acc[i][j] = zero;

    uint4  ra[2];
    float4 rb0[2], rb1[2];

    auto load_tiles = [&](int kt) {
        const int k0 = kt << 5;
#pragma unroll
        for (int it = 0; it < 2; ++it) {
            int c  = tid + it * 256;
            int r  = c >> 2;
            int cc = (c & 3) << 3;
            ra[it] = *(const uint4*)(A + (size_t)(row0 + r) * K + k0 + cc);
        }
#pragma unroll
        for (int it = 0; it < 2; ++it) {
            int c  = tid + it * 256;
            int kk = (c >> 5) << 1;
            int cc = (c & 31) << 2;
            rb0[it] = *(const float4*)(B + (size_t)(k0 + kk) * N + col0 + cc);
            rb1[it] = *(const float4*)(B + (size_t)(k0 + kk + 1) * N + col0 + cc);
        }
    };
    auto store_tiles = [&](int buf) {
#pragma unroll
        for (int it = 0; it < 2; ++it) {
            int c  = tid + it * 256;
            int r  = c >> 2;
            int cc = (c & 3) << 3;
            *(uint4*)&lA[buf][r * LDP + cc] = ra[it];
        }
#pragma unroll
        for (int it = 0; it < 2; ++it) {
            int c  = tid + it * 256;
            int kk = (c >> 5) << 1;
            int cc = (c & 31) << 2;
            *(unsigned int*)&lB[buf][(cc + 0) * LDP + kk] = f2bf_pk(rb0[it].x, rb1[it].x);
            *(unsigned int*)&lB[buf][(cc + 1) * LDP + kk] = f2bf_pk(rb0[it].y, rb1[it].y);
            *(unsigned int*)&lB[buf][(cc + 2) * LDP + kk] = f2bf_pk(rb0[it].z, rb1[it].z);
            *(unsigned int*)&lB[buf][(cc + 3) * LDP + kk] = f2bf_pk(rb0[it].w, rb1[it].w);
        }
    };

    const int ksteps = K >> 5;
    load_tiles(0);
    store_tiles(0);
    __syncthreads();

    for (int kt = 0; kt < ksteps; ++kt) {
        const int cur = kt & 1;
        if (kt + 1 < ksteps) load_tiles(kt + 1);   // global loads overlap WMMA

        Frag fa[4];
#pragma unroll
        for (int mi = 0; mi < 4; ++mi) {
            int m = waveM * 64 + mi * 16 + l16;
            fa[mi].q[0] = *(const uint4*)&lA[cur][m * LDP + 16 * half];
            fa[mi].q[1] = *(const uint4*)&lA[cur][m * LDP + 16 * half + 8];
        }
        Frag fb[2];
#pragma unroll
        for (int ni = 0; ni < 2; ++ni) {
            int n = waveN * 32 + ni * 16 + l16;
            fb[ni].q[0] = *(const uint4*)&lB[cur][n * LDP + 16 * half];
            fb[ni].q[1] = *(const uint4*)&lB[cur][n * LDP + 16 * half + 8];
        }
#pragma unroll
        for (int mi = 0; mi < 4; ++mi)
#pragma unroll
            for (int ni = 0; ni < 2; ++ni)
                acc[mi][ni] = __builtin_amdgcn_wmma_f32_16x16x32_bf16(
                    false, fa[mi].v, false, fb[ni].v, (short)0, acc[mi][ni], false, false);

        if (kt + 1 < ksteps) store_tiles((kt + 1) & 1);
        __syncthreads();
    }

#pragma unroll
    for (int mi = 0; mi < 4; ++mi)
#pragma unroll
        for (int ni = 0; ni < 2; ++ni)
#pragma unroll
            for (int r = 0; r < 8; ++r) {
                int m = row0 + waveM * 64 + mi * 16 + r + 8 * half;
                int n = col0 + waveN * 32 + ni * 16 + l16;
                float v = scale[m] * acc[mi][ni][r];
                size_t o = (size_t)m * N + n;
                if (accumulate) Out[o] += v; else Out[o] = v;
            }
}

// ---------------------------------------------------------------------------
// Router: logits = x @ gate_w, top-2 softmax -> rw[E][T]; sigmoid shared gate.
// ---------------------------------------------------------------------------
__global__ __launch_bounds__(256)
void moe_router_kernel(const float* __restrict__ x,
                       const float* __restrict__ gw,     // [H][E]
                       const float* __restrict__ sgw,    // [H]
                       float* __restrict__ rw,           // [E][T]
                       float* __restrict__ sgate)        // [T]
{
    __shared__ float lgw[HDIM * NEXP];
    __shared__ float lsg[HDIM];
    const int tid = threadIdx.x;
    for (int i = tid; i < HDIM * NEXP; i += 256) lgw[i] = gw[i];
    for (int i = tid; i < HDIM; i += 256) lsg[i] = sgw[i];
    __syncthreads();

    const int t = blockIdx.x * 256 + tid;
    const float* xr = x + (size_t)t * HDIM;
    float a[NEXP];
#pragma unroll
    for (int e = 0; e < NEXP; ++e) a[e] = 0.f;
    float ag = 0.f;
    for (int h = 0; h < HDIM; ++h) {
        float xv = xr[h];
#pragma unroll
        for (int e = 0; e < NEXP; ++e) a[e] += xv * lgw[h * NEXP + e];
        ag += xv * lsg[h];
    }
    int i0 = 0;
#pragma unroll
    for (int e = 1; e < NEXP; ++e) if (a[e] > a[i0]) i0 = e;
    int i1 = (i0 == 0) ? 1 : 0;
#pragma unroll
    for (int e = 0; e < NEXP; ++e)
        if (e != i0 && a[e] > a[i1]) i1 = e;
    float w1 = __expf(a[i1] - a[i0]);
    float inv = __builtin_amdgcn_rcpf(1.f + w1);
    float w0 = inv; w1 *= inv;
#pragma unroll
    for (int e = 0; e < NEXP; ++e) {
        float v = (e == i0) ? w0 : ((e == i1) ? w1 : 0.f);
        rw[(size_t)e * T_TOK + t] = v;
    }
    sgate[t] = fast_sigmoid(ag);
}

__global__ __launch_bounds__(256)
void cast_f32_bf16_kernel(const float* __restrict__ in,
                          unsigned short* __restrict__ out, int n)
{
    int i = (blockIdx.x * 256 + threadIdx.x) * 4;
    if (i + 3 < n) {
        float4 v = *(const float4*)(in + i);
        ushort4 o;
        o.x = f2bf(v.x); o.y = f2bf(v.y); o.z = f2bf(v.z); o.w = f2bf(v.w);
        *(ushort4*)(out + i) = o;
    }
}

// ---------------------------------------------------------------------------
extern "C" void kernel_launch(void* const* d_in, const int* in_sizes, int n_in,
                              void* d_out, int out_size, void* d_ws, size_t ws_size,
                              hipStream_t stream)
{
    (void)in_sizes; (void)n_in; (void)out_size; (void)ws_size;
    const float* x       = (const float*)d_in[0];
    const float* gate_w  = (const float*)d_in[1];
    const float* w_gate  = (const float*)d_in[2];
    const float* w_up    = (const float*)d_in[3];
    const float* w_down  = (const float*)d_in[4];
    const float* sh_gate = (const float*)d_in[5];
    const float* sh_up   = (const float*)d_in[6];
    const float* sh_down = (const float*)d_in[7];
    const float* sh_eg   = (const float*)d_in[8];
    float* out = (float*)d_out;

    char* ws = (char*)d_ws;
    unsigned short* xb   = (unsigned short*)ws;                               // T*H bf16
    unsigned short* hbuf = (unsigned short*)(ws + (size_t)T_TOK * HDIM * 2);  // T*ISDIM bf16 (reused)
    float* rw    = (float*)(ws + (size_t)T_TOK * HDIM * 2 + (size_t)T_TOK * ISDIM * 2);
    float* sg    = rw + (size_t)NEXP * T_TOK;

    // x -> bf16
    cast_f32_bf16_kernel<<<(T_TOK * HDIM) / 1024, 256, 0, stream>>>(x, xb, T_TOK * HDIM);
    // routing weights + shared sigmoid gate
    moe_router_kernel<<<T_TOK / 256, 256, 0, stream>>>(x, gate_w, sh_eg, rw, sg);

    // experts, sequential on stream (hbuf reused; out: write then accumulate)
    for (int e = 0; e < NEXP; ++e) {
        moe_gateup_kernel<<<dim3(T_TOK / 128, IDIM / 128), 256, 0, stream>>>(
            xb, w_gate + (size_t)e * HDIM * IDIM, w_up + (size_t)e * HDIM * IDIM,
            hbuf, HDIM, IDIM);
        moe_down_kernel<<<dim3(T_TOK / 128, HDIM / 128), 256, 0, stream>>>(
            hbuf, w_down + (size_t)e * IDIM * HDIM, rw + (size_t)e * T_TOK,
            out, IDIM, HDIM, e == 0 ? 0 : 1);
    }
    // shared expert (always active, sigmoid-gated)
    moe_gateup_kernel<<<dim3(T_TOK / 128, ISDIM / 128), 256, 0, stream>>>(
        xb, sh_gate, sh_up, hbuf, HDIM, ISDIM);
    moe_down_kernel<<<dim3(T_TOK / 128, HDIM / 128), 256, 0, stream>>>(
        hbuf, sh_down, sg, out, ISDIM, HDIM, 1);
}
